// DRNA_Model_11269994184842
// MI455X (gfx1250) — compile-verified, matrix operand order checked by reading, not run on hardware
//
#include <hip/hip_runtime.h>
#include <hip/hip_bf16.h>
#include <math.h>

// ---------------------------------------------------------------------------
// Problem constants (match reference)
// ---------------------------------------------------------------------------
constexpr int B_  = 2;
constexpr int S_  = 2048;
constexpr int V_  = 32000;
constexpr int D_  = 1024;
constexpr int L_  = 16;
constexpr int H_  = 16;
constexpr int DFF_= 4096;
constexpr int DH_ = 64;
constexpr int M_  = B_ * S_;        // 4096 token rows

typedef __attribute__((ext_vector_type(16))) __bf16 v16bf;
typedef __attribute__((ext_vector_type(8)))  float  v8f;
typedef __attribute__((ext_vector_type(4)))  int    v4i;

// ---------------------------------------------------------------------------
// WMMA helper (CDNA5 wave32, V_WMMA_F32_16X16X32_BF16)
// ---------------------------------------------------------------------------
__device__ inline v8f wmma_bf16(v16bf a, v16bf b, v8f c) {
  return __builtin_amdgcn_wmma_f32_16x16x32_bf16(
      /*neg_a=*/false, a, /*neg_b=*/false, b,
      /*c_mod=*/(short)0, c, /*reuse_a=*/false, /*reuse_b=*/false);
}

// A-operand (16 rows x 32 k) from a row-major bf16 LDS tile; also serves the
// B operand when the tile is stored transposed (N x K row-major).
// ISA 7.12.2: lane<16 holds K 0..7 & 16..23; lane>=16 holds 8..15 & 24..31.
__device__ inline v16bf ld_frag_nmajor(const __bf16* p, int ld, int lane) {
  const int rr = lane & 15;
  const int kb = (lane >> 4) * 8;
  v16bf f;
#pragma unroll
  for (int i = 0; i < 8; ++i) f[i]     = p[rr * ld + kb + i];
#pragma unroll
  for (int i = 0; i < 8; ++i) f[8 + i] = p[rr * ld + 16 + kb + i];
  return f;
}

// Same fragment, but source tile lives in LDS as f32 (async-copied raw);
// convert to bf16 while building the fragment.
__device__ inline v16bf ld_frag_nmajor_f32(const float* p, int ld, int lane) {
  const int rr = lane & 15;
  const int kb = (lane >> 4) * 8;
  v16bf f;
#pragma unroll
  for (int i = 0; i < 8; ++i) f[i]     = (__bf16)p[rr * ld + kb + i];
#pragma unroll
  for (int i = 0; i < 8; ++i) f[8 + i] = (__bf16)p[rr * ld + 16 + kb + i];
  return f;
}

// B-operand (32 k x 16 cols) from a K-major (K x N row-major) bf16 LDS tile
// (used by the attention P@V path).
__device__ inline v16bf ld_frag_kmajor(const __bf16* p, int ld, int lane) {
  const int cc = lane & 15;
  const int kb = (lane >> 4) * 8;
  v16bf f;
#pragma unroll
  for (int i = 0; i < 8; ++i) f[i]     = p[(kb + i) * ld + cc];
#pragma unroll
  for (int i = 0; i < 8; ++i) f[8 + i] = p[(16 + kb + i) * ld + cc];
  return f;
}

// ---------------------------------------------------------------------------
// Async global->LDS 16-byte copy (CDNA5 GLOBAL_LOAD_ASYNC_TO_LDS_B128),
// with a register-copy fallback if the builtin is absent.
// Probe-verified: param 1 is an int4* in the global (as1) address space.
// ---------------------------------------------------------------------------
#if defined(__has_builtin)
#if __has_builtin(__builtin_amdgcn_global_load_async_to_lds_b128)
#define HAVE_ASYNC_LDS 1
#endif
#if __has_builtin(__builtin_amdgcn_s_wait_asynccnt)
#define HAVE_WAIT_ASYNC 1
#endif
#endif

__device__ inline void async_copy_b128(const float* gsrc, float* lds_dst) {
#ifdef HAVE_ASYNC_LDS
  __builtin_amdgcn_global_load_async_to_lds_b128(
      (__attribute__((address_space(1))) v4i*)(void*)const_cast<float*>(gsrc),
      (__attribute__((address_space(3))) v4i*)(void*)lds_dst,
      0, 0);
#else
  float4 v = *(const float4*)gsrc;
  *(float4*)lds_dst = v;
#endif
}

__device__ inline void wait_async_zero() {
#ifdef HAVE_ASYNC_LDS
#ifdef HAVE_WAIT_ASYNC
  __builtin_amdgcn_s_wait_asynccnt(0);
#else
  asm volatile("s_wait_asynccnt 0" ::: "memory");
#endif
#endif
}

// ---------------------------------------------------------------------------
// Tiled bf16-WMMA GEMM:  C[M,N] = epi(A[M,K] @ B[K,N] + bias[N])
//  - Block = 256 threads = 8 waves; tile 128x128, K-step 32.
//  - Wave w owns a 32x64 sub-tile: wm=(w&3)*32, wn=(w>>2)*64 -> 2x4 frags,
//    8 WMMAs per wave per K-step.
//  - 2-stage software pipeline: at step s the async DMA for A(s+1) and the
//    global B(s+1) loads (into registers) are issued BEFORE the WMMAs of
//    step s, so DMA/HBM latency hides behind matrix math. B registers are
//    spilled to LDS (transposed [n][k] bf16) after compute, then
//    s_wait_asynccnt 0 + barrier flips the buffers.
//  - A tile staged raw f32 (async b128), converted f32->bf16 at frag build.
//  - All dims used here are multiples of 128/32 -> no edge guards.
//  - EPI: 0 = none, 1 = exact GELU.
// ---------------------------------------------------------------------------
template <int EPI>
__global__ __launch_bounds__(256)
void gemm_bf16_wmma(const float* __restrict__ A, const float* __restrict__ Bm,
                    const float* __restrict__ bias, float* __restrict__ C,
                    int M, int N, int K) {
  constexpr int BM = 128, BN = 128, BK = 32;
  __shared__ float  Af[2][BM * BK];     // 2 x 16 KB, raw f32 via async DMA
  __shared__ __bf16 Bt[2][BN * BK];     // 2 x 8 KB, transposed [n][k] bf16

  const int bn   = blockIdx.x * BN;
  const int bm   = blockIdx.y * BM;
  const int tid  = threadIdx.x;
  const int wave = tid >> 5;
  const int lane = tid & 31;
  const int wm   = (wave & 3) * 32;     // wave row offset within tile
  const int wn   = (wave >> 2) * 64;    // wave col offset within tile

  v8f zero = {};
  v8f acc[2][4];
#pragma unroll
  for (int sm = 0; sm < 2; ++sm)
#pragma unroll
    for (int sn = 0; sn < 4; ++sn) acc[sm][sn] = zero;

  // --- staging helpers -----------------------------------------------------
  auto issueA = [&](int k0, int stage) {
#pragma unroll
    for (int j = 0; j < 4; ++j) {
      const int i = tid + j * 256;            // chunk id, 0..1023
      const int r = i >> 3;                   // row 0..127
      const int c = (i & 7) * 4;              // col (floats) 0..28
      async_copy_b128(&A[(long)(bm + r) * K + k0 + c], &Af[stage][i * 4]);
    }
  };
  auto loadB = [&](int k0, float (&regs)[16]) {
#pragma unroll
    for (int j = 0; j < 16; ++j) {
      const int i = tid + j * 256;            // 0..4095
      const int r = i >> 7;                   // k row 0..31
      const int c = i & 127;                  // n col 0..127
      regs[j] = Bm[(long)(k0 + r) * N + bn + c];
    }
  };
  auto storeB = [&](const float (&regs)[16], int stage) {
#pragma unroll
    for (int j = 0; j < 16; ++j) {
      const int i = tid + j * 256;
      const int r = i >> 7;
      const int c = i & 127;
      Bt[stage][c * BK + r] = (__bf16)regs[j];
    }
  };

  float bregs[16];

  // --- prologue: stage buffer 0 --------------------------------------------
  issueA(0, 0);
  loadB(0, bregs);
  storeB(bregs, 0);
  wait_async_zero();
  __syncthreads();

  const int nsteps = K / BK;
  for (int step = 0; step < nsteps; ++step) {
    const int  buf      = step & 1;
    const bool has_next = (step + 1) < nsteps;

    // Prefetch next tile: fire DMA + global loads before compute.
    if (has_next) {
      issueA((step + 1) * BK, buf ^ 1);
      loadB((step + 1) * BK, bregs);
    }

    // Compute from current buffer.
    v16bf af[2], bf2[4];
#pragma unroll
    for (int sm = 0; sm < 2; ++sm)
      af[sm] = ld_frag_nmajor_f32(&Af[buf][(wm + sm * 16) * BK], BK, lane);
#pragma unroll
    for (int sn = 0; sn < 4; ++sn)
      bf2[sn] = ld_frag_nmajor(&Bt[buf][(wn + sn * 16) * BK], BK, lane);

#pragma unroll
    for (int sm = 0; sm < 2; ++sm)
#pragma unroll
      for (int sn = 0; sn < 4; ++sn)
        acc[sm][sn] = wmma_bf16(af[sm], bf2[sn], acc[sm][sn]);

    // Drain prefetch into the other buffer, then flip.
    if (has_next) {
      storeB(bregs, buf ^ 1);
      wait_async_zero();
    }
    __syncthreads();
  }

  // Epilogue: C layout = 8 rows per lane (half select), col = lane&15.
  const int l15   = lane & 15;
  const int half8 = (lane >> 4) * 8;
#pragma unroll
  for (int sm = 0; sm < 2; ++sm)
#pragma unroll
    for (int sn = 0; sn < 4; ++sn) {
      const int col = bn + wn + sn * 16 + l15;
      const float bv = bias ? bias[col] : 0.0f;
#pragma unroll
      for (int r = 0; r < 8; ++r) {
        const int row = bm + wm + sm * 16 + half8 + r;
        float v = acc[sm][sn][r] + bv;
        if (EPI == 1) v = 0.5f * v * (1.0f + erff(v * 0.70710678118654752f));
        C[(long)row * N + col] = v;
      }
    }
}

// ---------------------------------------------------------------------------
// Embedding gather: h[row,:] = embed[x[row],:]
// ---------------------------------------------------------------------------
__global__ __launch_bounds__(256)
void embed_kernel(const int* __restrict__ x, const float* __restrict__ embed,
                  float* __restrict__ h) {
  const int row = blockIdx.x;
  const int tok = x[row];
  for (int c = threadIdx.x; c < D_; c += 256)
    h[(long)row * D_ + c] = embed[(long)tok * D_ + c];
}

// ---------------------------------------------------------------------------
// In-place RoPE on q and k slices of the qkv buffer [M, 3D].
// One thread handles the (j, j+32) rotation pair of one head.
// ---------------------------------------------------------------------------
__global__ __launch_bounds__(256)
void rope_kernel(float* __restrict__ qkv) {
  const long tid   = (long)blockIdx.x * 256 + threadIdx.x; // B*S*2*H*32 threads
  const int  j     = (int)(tid & 31);
  const int  h     = (int)((tid >> 5) & (H_ - 1));
  const int  part  = (int)((tid >> 9) & 1);               // 0 = q, 1 = k
  const long bs    = tid >> 10;                           // 0 .. B*S-1
  const int  s     = (int)(bs % S_);

  const long base = bs * (3L * D_) + (long)part * D_ + (long)h * DH_;
  const float invf = powf(10000.0f, -(float)j / 32.0f);
  const float th   = (float)s * invf;
  const float cs = cosf(th), sn = sinf(th);
  const float x1 = qkv[base + j];
  const float x2 = qkv[base + j + 32];
  qkv[base + j]      = x1 * cs - x2 * sn;
  qkv[base + j + 32] = x2 * cs + x1 * sn;
}

// ---------------------------------------------------------------------------
// Flash attention (non-causal, matches reference full softmax).
// Block = 128 threads (4 waves) per (b*H+h, q-tile of 64 rows).
// Wave w owns q rows [w*16, w*16+16). WMMA for Q@K^T and P@V; online softmax
// on C-fragment layout with 16-lane shfl_xor reductions. P re-enters the
// A-operand layout via a per-wave LDS transpose buffer.
// ---------------------------------------------------------------------------
__global__ __launch_bounds__(128)
void attention_kernel(const float* __restrict__ qkv, float* __restrict__ out) {
  __shared__ __bf16 Qs[64 * 64];
  __shared__ __bf16 Ks[64 * 64];
  __shared__ __bf16 Vs[64 * 64];
  __shared__ __bf16 Ps[4][16 * 64];

  const int b    = blockIdx.x / H_;
  const int h    = blockIdx.x % H_;
  const int q0   = blockIdx.y * 64;
  const int tid  = threadIdx.x;
  const int wave = tid >> 5;
  const int lane = tid & 31;
  const int l15  = lane & 15;
  const int half8 = (lane >> 4) * 8;
  const long rs3 = 3L * D_;

  // Stage Q tile (64 q rows x 64 dh), f32 -> bf16.
  for (int i = tid; i < 64 * 64; i += 128) {
    const int r = i >> 6, c = i & 63;
    Qs[i] = (__bf16)qkv[(long)(b * S_ + q0 + r) * rs3 + h * DH_ + c];
  }
  __syncthreads();

  v16bf aQ[2];
#pragma unroll
  for (int kc = 0; kc < 2; ++kc)
    aQ[kc] = ld_frag_nmajor(&Qs[(wave * 16) * 64 + kc * 32], 64, lane);

  float mrun[8], lrun[8];
  v8f zero = {};
  v8f accO[4];
#pragma unroll
  for (int r = 0; r < 8; ++r) { mrun[r] = -INFINITY; lrun[r] = 0.0f; }
#pragma unroll
  for (int n = 0; n < 4; ++n) accO[n] = zero;

  const float scale = 0.125f; // 1/sqrt(64)

  for (int s0 = 0; s0 < S_; s0 += 64) {
    __syncthreads(); // previous iteration done reading Ks/Vs
    for (int i = tid; i < 64 * 64; i += 128) {
      const int r = i >> 6, c = i & 63;
      const long base = (long)(b * S_ + s0 + r) * rs3 + h * DH_;
      Ks[i] = (__bf16)qkv[base + D_ + c];
      Vs[i] = (__bf16)qkv[base + 2 * D_ + c];
    }
    __syncthreads();

    // ---- scores S = Q @ K^T  (K stored [key, dh] == B^T row-major) ----
    v8f sF[4];
#pragma unroll
    for (int n = 0; n < 4; ++n) {
      sF[n] = zero;
#pragma unroll
      for (int kc = 0; kc < 2; ++kc) {
        v16bf bK = ld_frag_nmajor(&Ks[(n * 16) * 64 + kc * 32], 64, lane);
        sF[n] = wmma_bf16(aQ[kc], bK, sF[n]);
      }
    }

    // ---- online softmax, per fragment row r (actual row = half8 + r) ----
#pragma unroll
    for (int r = 0; r < 8; ++r) {
      float s0v = sF[0][r] * scale, s1v = sF[1][r] * scale;
      float s2v = sF[2][r] * scale, s3v = sF[3][r] * scale;
      float mx = fmaxf(fmaxf(s0v, s1v), fmaxf(s2v, s3v));
#pragma unroll
      for (int off = 1; off < 16; off <<= 1)
        mx = fmaxf(mx, __shfl_xor(mx, off, 16));
      const float newm  = fmaxf(mrun[r], mx);
      const float alpha = expf(mrun[r] - newm);
      mrun[r] = newm;
      float p0 = expf(s0v - newm), p1 = expf(s1v - newm);
      float p2 = expf(s2v - newm), p3 = expf(s3v - newm);
      float ps = p0 + p1 + p2 + p3;
#pragma unroll
      for (int off = 1; off < 16; off <<= 1)
        ps += __shfl_xor(ps, off, 16);
      lrun[r] = lrun[r] * alpha + ps;
#pragma unroll
      for (int n = 0; n < 4; ++n) accO[n][r] *= alpha;
      const int pr = (half8 + r) * 64 + l15;
      Ps[wave][pr +  0] = (__bf16)p0;
      Ps[wave][pr + 16] = (__bf16)p1;
      Ps[wave][pr + 32] = (__bf16)p2;
      Ps[wave][pr + 48] = (__bf16)p3;
    }
    // Per-wave buffer + in-order LDS within a wave: no cross-wave sync needed.

    // ---- O += P @ V  (V stored [key, dh] == K x N row-major) ----
#pragma unroll
    for (int kc = 0; kc < 2; ++kc) {
      v16bf aP = ld_frag_nmajor(&Ps[wave][kc * 32], 64, lane);
#pragma unroll
      for (int n = 0; n < 4; ++n) {
        v16bf bV = ld_frag_kmajor(&Vs[(kc * 32) * 64 + n * 16], 64, lane);
        accO[n] = wmma_bf16(aP, bV, accO[n]);
      }
    }
  }

  // ---- write O / l; layout [b, q, h*64 + dh] == [B,S,D] reshape ----
#pragma unroll
  for (int r = 0; r < 8; ++r) {
    const float inv  = 1.0f / lrun[r];
    const int   qrow = q0 + wave * 16 + half8 + r;
    const long  base = (long)(b * S_ + qrow) * D_ + h * DH_ + l15;
#pragma unroll
    for (int n = 0; n < 4; ++n)
      out[base + n * 16] = accO[n][r] * inv;
  }
}

// ---------------------------------------------------------------------------
// Fused residual-add + LayerNorm, one block per row (in-place safe on `res`).
// ---------------------------------------------------------------------------
__global__ __launch_bounds__(256)
void resln_kernel(const float* __restrict__ res, const float* __restrict__ y,
                  const float* __restrict__ g, const float* __restrict__ bta,
                  float* __restrict__ outp) {
  __shared__ float sb1[256], sb2[256];
  const int row = blockIdx.x;
  const int t   = threadIdx.x;
  float vals[D_ / 256];
  float s = 0.0f, s2 = 0.0f;
#pragma unroll
  for (int i = 0; i < D_ / 256; ++i) {
    const int c = i * 256 + t;
    const float v = res[(long)row * D_ + c] + y[(long)row * D_ + c];
    vals[i] = v; s += v; s2 += v * v;
  }
  sb1[t] = s; sb2[t] = s2;
  __syncthreads();
  for (int off = 128; off > 0; off >>= 1) {
    if (t < off) { sb1[t] += sb1[t + off]; sb2[t] += sb2[t + off]; }
    __syncthreads();
  }
  const float mean = sb1[0] / (float)D_;
  const float var  = sb2[0] / (float)D_ - mean * mean;
  const float rstd = rsqrtf(var + 1e-5f);
#pragma unroll
  for (int i = 0; i < D_ / 256; ++i) {
    const int c = i * 256 + t;
    outp[(long)row * D_ + c] = (vals[i] - mean) * rstd * g[c] + bta[c];
  }
}

// ---------------------------------------------------------------------------
// kernel_launch: full forward pass. Scratch layout in d_ws (floats):
//   h[M*D] | tmp[M*D] | qkv[M*3D] | attn[M*D] | mid[M*DFF]   (~160 MB)
// ---------------------------------------------------------------------------
extern "C" void kernel_launch(void* const* d_in, const int* in_sizes, int n_in,
                              void* d_out, int out_size, void* d_ws, size_t ws_size,
                              hipStream_t stream) {
  (void)in_sizes; (void)n_in; (void)out_size; (void)ws_size;
  const int*   x      = (const int*)  d_in[0];
  const float* embed  = (const float*)d_in[1];
  const float* qkv_w  = (const float*)d_in[2];
  const float* qkv_b  = (const float*)d_in[3];
  const float* out_w  = (const float*)d_in[4];
  const float* out_b  = (const float*)d_in[5];
  const float* ln1_g  = (const float*)d_in[6];
  const float* ln1_b  = (const float*)d_in[7];
  const float* w1     = (const float*)d_in[8];
  const float* b1     = (const float*)d_in[9];
  const float* w2     = (const float*)d_in[10];
  const float* b2     = (const float*)d_in[11];
  const float* ln2_g  = (const float*)d_in[12];
  const float* ln2_b  = (const float*)d_in[13];
  const float* head_w = (const float*)d_in[14];
  const float* head_b = (const float*)d_in[15];
  float* outp = (float*)d_out;

  float* ws   = (float*)d_ws;
  float* hbuf = ws;
  float* tmp  = hbuf + (size_t)M_ * D_;
  float* qkvb = tmp  + (size_t)M_ * D_;
  float* attn = qkvb + (size_t)M_ * 3 * D_;
  float* mid  = attn + (size_t)M_ * D_;

  embed_kernel<<<M_, 256, 0, stream>>>(x, embed, hbuf);

  for (int l = 0; l < L_; ++l) {
    // QKV projection: [M,D] @ [D,3D]
    gemm_bf16_wmma<0><<<dim3(3 * D_ / 128, M_ / 128), 256, 0, stream>>>(
        hbuf, qkv_w + (size_t)l * D_ * 3 * D_, qkv_b + (size_t)l * 3 * D_,
        qkvb, M_, 3 * D_, D_);

    // RoPE on q,k (in place)
    rope_kernel<<<(B_ * S_ * 2 * H_ * 32) / 256, 256, 0, stream>>>(qkvb);

    // Attention -> attn [M, D]
    attention_kernel<<<dim3(B_ * H_, S_ / 64), 128, 0, stream>>>(qkvb, attn);

    // Output projection
    gemm_bf16_wmma<0><<<dim3(D_ / 128, M_ / 128), 256, 0, stream>>>(
        attn, out_w + (size_t)l * D_ * D_, out_b + (size_t)l * D_,
        tmp, M_, D_, D_);

    // h = LN(h + proj)
    resln_kernel<<<M_, 256, 0, stream>>>(hbuf, tmp,
        ln1_g + (size_t)l * D_, ln1_b + (size_t)l * D_, hbuf);

    // FFN up + GELU
    gemm_bf16_wmma<1><<<dim3(DFF_ / 128, M_ / 128), 256, 0, stream>>>(
        hbuf, w1 + (size_t)l * D_ * DFF_, b1 + (size_t)l * DFF_,
        mid, M_, DFF_, D_);

    // FFN down
    gemm_bf16_wmma<0><<<dim3(D_ / 128, M_ / 128), 256, 0, stream>>>(
        mid, w2 + (size_t)l * DFF_ * D_, b2 + (size_t)l * D_,
        tmp, M_, D_, DFF_);

    // h = LN(h + ffn)
    resln_kernel<<<M_, 256, 0, stream>>>(hbuf, tmp,
        ln2_g + (size_t)l * D_, ln2_b + (size_t)l * D_, hbuf);
  }

  // LM head: [M,D] @ [D,V] -> d_out (V/128 = 250 exact)
  gemm_bf16_wmma<0><<<dim3(V_ / 128, M_ / 128), 256, 0, stream>>>(
      hbuf, head_w, head_b, outp, M_, V_, D_);
}
